// FasterRCNN_41944650612767
// MI455X (gfx1250) — matrix operand order repaired and checked by reading, not compiled, not run on hardware
//
#include <hip/hip_runtime.h>
#include <hip/hip_bf16.h>
#include <math.h>

// ---------------------------------------------------------------------------
// Types for CDNA5 WMMA (wave32): 16x16x32 bf16 -> f32
// ---------------------------------------------------------------------------
typedef __attribute__((ext_vector_type(16))) __bf16 v16bf;
typedef __attribute__((ext_vector_type(8)))  float  v8f;
typedef __attribute__((ext_vector_type(4)))  int    v4i;

typedef __attribute__((address_space(1))) v4i g_v4i;   // global 16B chunk
typedef __attribute__((address_space(3))) v4i l_v4i;   // LDS 16B chunk

#if __has_builtin(__builtin_amdgcn_global_load_async_to_lds_b128) && \
    __has_builtin(__builtin_amdgcn_s_wait_asynccnt)
#define HAVE_ASYNC_LDS 1
#endif

#define FH_ 37
#define FW_ 50
#define N_ANCHOR_ 9
#define N_ANCH_TOT (FH_ * FW_ * N_ANCHOR_)   // 16650
#define PRE_NMS_ 6000
#define POST_NMS_ 300
#define SORT_N 32768
#define NMS_THRESH_ 0.7f
#define IMG_H_ 600.0f
#define IMG_W_ 800.0f

// Stage one 16(M) x 32(K) f32 tile of a row-major [Mtot][K] matrix into LDS.
// 128 threads, each moves 16 bytes -> 2KB tile in one async b128 per lane.
__device__ __forceinline__ void stage_a_tile(const float* __restrict__ src_mat,
                                             float* __restrict__ dst,   // 512 floats
                                             int m0, int k0, int Mtot, int K)
{
    int t   = threadIdx.x;          // 0..127
    int row = t >> 3;               // 16 rows
    int col = (t & 7) * 4;          // 8 x 4 floats
    int mr  = m0 + row; if (mr >= Mtot) mr = Mtot - 1;
    int kk  = k0 + col; if (kk + 4 > K) kk = 0;      // keep reads in-bounds; masked later
    const float* src = src_mat + (size_t)mr * K + kk;
    float* d = dst + row * 32 + col;
#ifdef HAVE_ASYNC_LDS
    __builtin_amdgcn_global_load_async_to_lds_b128(
        (g_v4i*)(v4i*)(float*)src, (l_v4i*)(v4i*)d, 0, 0);
#else
    d[0] = src[0]; d[1] = src[1]; d[2] = src[2]; d[3] = src[3];
#endif
}

__device__ __forceinline__ void wait_stage()
{
#ifdef HAVE_ASYNC_LDS
    __builtin_amdgcn_s_wait_asynccnt(0);
#endif
}

// ---------------------------------------------------------------------------
// Implicit-GEMM convolution: out[Cout][H*W] = relu(W[Cout][Cin*ks*ks] * im2col + b)
// Block = 128 threads (4 waves). Block tile: 16(M=Cout) x 256(N=pixels);
// each wave owns a 16x64 tile -> 4 WMMAs per K-chunk sharing one A fragment.
// A (weights) tile is staged block-wide into LDS via async-to-LDS, double buffered.
// ---------------------------------------------------------------------------
__global__ void conv_wmma(const float* __restrict__ in, const float* __restrict__ wgt,
                          const float* __restrict__ bias, float* __restrict__ out,
                          int Cin, int Cout, int H, int W, int ks, int relu)
{
    __shared__ float sA[2][16 * 32];

    const int lane = threadIdx.x & 31;
    const int wave = threadIdx.x >> 5;
    const int HW   = H * W;
    const int K    = Cin * ks * ks;
    const int m0   = blockIdx.y * 16;
    const int n0   = blockIdx.x * 256 + wave * 64;
    const int half = lane >> 4;
    const int l16  = lane & 15;
    const int mIdx = m0 + l16;

    int nIdx[4], oy[4], ox[4];
#pragma unroll
    for (int s = 0; s < 4; ++s) {
        nIdx[s] = n0 + s * 16 + l16;
        oy[s]   = (nIdx[s] < HW) ? (nIdx[s] / W) : 0;
        ox[s]   = (nIdx[s] < HW) ? (nIdx[s] - oy[s] * W) : 0;
    }

    const int nk = (K + 31) / 32;
    stage_a_tile(wgt, sA[0], m0, 0, Cout, K);

    v8f acc[4] = {};
    for (int kc = 0; kc < nk; ++kc) {
        const int kb = kc & 1;
        const int k0 = kc * 32;
        wait_stage();
        __syncthreads();
        if (kc + 1 < nk)
            stage_a_tile(wgt, sA[kb ^ 1], m0, k0 + 32, Cout, K);

        // ---- A fragment from LDS (bf16, contiguous k = e + 16*half, same order as B) ----
        v16bf afrag;
#pragma unroll
        for (int e = 0; e < 16; ++e) {
            int kl = e + 16 * half;     // 0..31
            float v = (mIdx < Cout && (k0 + kl) < K) ? sA[kb][l16 * 32 + kl] : 0.0f;
            afrag[e] = (__bf16)v;
        }

        // ---- 4 B fragments (im2col), 4 WMMAs sharing afrag ----
#pragma unroll
        for (int s = 0; s < 4; ++s) {
            v16bf bfrag;
#pragma unroll
            for (int e = 0; e < 16; ++e) {
                int k = k0 + e + 16 * half;
                float v = 0.0f;
                if (nIdx[s] < HW && k < K) {
                    int cin, iy, ix;
                    if (ks == 3) {
                        cin = k / 9;
                        int r  = k - cin * 9;
                        int ky = r / 3;
                        int kx = r - ky * 3;
                        iy = oy[s] + ky - 1;
                        ix = ox[s] + kx - 1;
                    } else {
                        cin = k; iy = oy[s]; ix = ox[s];
                    }
                    if (iy >= 0 && iy < H && ix >= 0 && ix < W)
                        v = in[(size_t)cin * HW + iy * W + ix];
                }
                bfrag[e] = (__bf16)v;
            }
            acc[s] = __builtin_amdgcn_wmma_f32_16x16x32_bf16(
                         false, afrag, false, bfrag, (short)0, acc[s], false, false);
        }
    }

    // ---- store D: VGPR r -> M = r + 8*half, N = lane&15 ----
#pragma unroll
    for (int s = 0; s < 4; ++s) {
#pragma unroll
        for (int r = 0; r < 8; ++r) {
            int m = m0 + r + 8 * half;
            if (m < Cout && nIdx[s] < HW) {
                float v = acc[s][r] + bias[m];
                if (relu) v = fmaxf(v, 0.0f);
                out[(size_t)m * HW + nIdx[s]] = v;
            }
        }
    }
}

// ---------------------------------------------------------------------------
// FC GEMM: out[M][N] = act(A[M][K] * W[N][K]^T + b)
// Same blocking: block tile 16(M) x 256(N); A tile staged via async-to-LDS.
// ---------------------------------------------------------------------------
__global__ void fc_wmma(const float* __restrict__ A, const float* __restrict__ Wt,
                        const float* __restrict__ bias, float* __restrict__ out,
                        int M, int N, int K, int relu)
{
    __shared__ float sA[2][16 * 32];

    const int lane = threadIdx.x & 31;
    const int wave = threadIdx.x >> 5;
    const int m0   = blockIdx.y * 16;
    const int n0   = blockIdx.x * 256 + wave * 64;
    const int half = lane >> 4;
    const int l16  = lane & 15;
    const int mIdx = m0 + l16;

    const float* wrow[4];
    int nIdx[4];
#pragma unroll
    for (int s = 0; s < 4; ++s) {
        nIdx[s] = n0 + s * 16 + l16;
        wrow[s] = Wt + (size_t)((nIdx[s] < N) ? nIdx[s] : 0) * K;
    }

    const int nk = (K + 31) / 32;
    stage_a_tile(A, sA[0], m0, 0, M, K);

    v8f acc[4] = {};
    for (int kc = 0; kc < nk; ++kc) {
        const int kb = kc & 1;
        const int k0 = kc * 32;
        wait_stage();
        __syncthreads();
        if (kc + 1 < nk)
            stage_a_tile(A, sA[kb ^ 1], m0, k0 + 32, M, K);

        v16bf afrag;
#pragma unroll
        for (int e = 0; e < 16; ++e) {
            int kl = e + 16 * half;
            float v = (mIdx < M && (k0 + kl) < K) ? sA[kb][l16 * 32 + kl] : 0.0f;
            afrag[e] = (__bf16)v;
        }

#pragma unroll
        for (int s = 0; s < 4; ++s) {
            if (nIdx[s] < N) __builtin_prefetch(wrow[s] + k0 + 64, 0, 1);
            v16bf bfrag;
#pragma unroll
            for (int e = 0; e < 16; ++e) {
                int k = k0 + e + 16 * half;
                float v = (nIdx[s] < N && k < K) ? wrow[s][k] : 0.0f;
                bfrag[e] = (__bf16)v;
            }
            acc[s] = __builtin_amdgcn_wmma_f32_16x16x32_bf16(
                         false, afrag, false, bfrag, (short)0, acc[s], false, false);
        }
    }

#pragma unroll
    for (int s = 0; s < 4; ++s) {
#pragma unroll
        for (int r = 0; r < 8; ++r) {
            int m = m0 + r + 8 * half;
            if (m < M && nIdx[s] < N) {
                float v = acc[s][r] + bias[nIdx[s]];
                if (relu) v = fmaxf(v, 0.0f);
                out[(size_t)m * N + nIdx[s]] = v;
            }
        }
    }
}

// ---------------------------------------------------------------------------
// 2x2 max-pool, stride 2, VALID
// ---------------------------------------------------------------------------
__global__ void maxpool2(const float* __restrict__ in, float* __restrict__ out,
                         int C, int H, int W)
{
    int Ho = H >> 1, Wo = W >> 1;
    size_t total = (size_t)C * Ho * Wo;
    size_t t = (size_t)blockIdx.x * blockDim.x + threadIdx.x;
    if (t >= total) return;
    int wo = t % Wo;
    int ho = (t / Wo) % Ho;
    int c  = t / ((size_t)Wo * Ho);
    const float* p = in + (size_t)c * H * W + (size_t)(2 * ho) * W + 2 * wo;
    float v = fmaxf(fmaxf(p[0], p[1]), fmaxf(p[W], p[W + 1]));
    out[t] = v;
}

// ---------------------------------------------------------------------------
// Sort key/index init (padded to SORT_N with -inf)
// ---------------------------------------------------------------------------
__global__ void sort_init(float* __restrict__ keys, int* __restrict__ idxs)
{
    int i = blockIdx.x * blockDim.x + threadIdx.x;
    if (i < SORT_N) { keys[i] = -INFINITY; idxs[i] = i; }
}

// ---------------------------------------------------------------------------
// Anchor decode + softmax fg + clip + min-size mask -> boxes, keys
// ---------------------------------------------------------------------------
__global__ void proposal_prep(const float* __restrict__ rpnloc,
                              const float* __restrict__ rpnscore,
                              float* __restrict__ boxes, float* __restrict__ keys)
{
    int t = blockIdx.x * blockDim.x + threadIdx.x;
    if (t >= N_ANCH_TOT) return;
    const int HW = FH_ * FW_;
    int a = t % N_ANCHOR_;
    int p = t / N_ANCHOR_;
    int y = p / FW_;
    int x = p - y * FW_;

    float l0 = rpnloc[(size_t)(a * 4 + 0) * HW + p];
    float l1 = rpnloc[(size_t)(a * 4 + 1) * HW + p];
    float l2 = rpnloc[(size_t)(a * 4 + 2) * HW + p];
    float l3 = rpnloc[(size_t)(a * 4 + 3) * HW + p];
    float s0 = rpnscore[(size_t)(a * 2 + 0) * HW + p];
    float s1 = rpnscore[(size_t)(a * 2 + 1) * HW + p];
    float fg = 1.0f / (1.0f + expf(s0 - s1));

    const float ratios[3] = {0.5f, 1.0f, 2.0f};
    const float scales[3] = {8.0f, 16.0f, 32.0f};
    float ratio = ratios[a / 3];
    float scale = scales[a % 3];
    float hs = 16.0f * scale * sqrtf(ratio);
    float wz = 16.0f * scale * sqrtf(1.0f / ratio);
    float ay1 = 8.0f - 0.5f * hs + 16.0f * (float)y;
    float ax1 = 8.0f - 0.5f * wz + 16.0f * (float)x;
    float ay2 = 8.0f + 0.5f * hs + 16.0f * (float)y;
    float ax2 = 8.0f + 0.5f * wz + 16.0f * (float)x;

    float h  = ay2 - ay1, w = ax2 - ax1;
    float cy = ay1 + 0.5f * h, cx = ax1 + 0.5f * w;
    float ncy = l0 * h + cy, ncx = l1 * w + cx;
    float nh  = expf(l2) * h, nw = expf(l3) * w;

    float y1 = fminf(fmaxf(ncy - 0.5f * nh, 0.0f), IMG_H_);
    float x1 = fminf(fmaxf(ncx - 0.5f * nw, 0.0f), IMG_W_);
    float y2 = fminf(fmaxf(ncy + 0.5f * nh, 0.0f), IMG_H_);
    float x2 = fminf(fmaxf(ncx + 0.5f * nw, 0.0f), IMG_W_);

    boxes[t * 4 + 0] = y1; boxes[t * 4 + 1] = x1;
    boxes[t * 4 + 2] = y2; boxes[t * 4 + 3] = x2;
    bool keep = ((y2 - y1) >= 16.0f) && ((x2 - x1) >= 16.0f);
    keys[t] = keep ? fg : -INFINITY;
}

// ---------------------------------------------------------------------------
// Bitonic sort step (descending overall), key+payload
// ---------------------------------------------------------------------------
__global__ void bitonic_step(float* __restrict__ keys, int* __restrict__ idxs,
                             int j, int k)
{
    int i = blockIdx.x * blockDim.x + threadIdx.x;
    int p = i ^ j;
    if (p > i && i < SORT_N) {
        float ki = keys[i], kp = keys[p];
        bool desc = ((i & k) == 0);
        bool sw = desc ? (ki < kp) : (ki > kp);
        if (sw) {
            keys[i] = kp; keys[p] = ki;
            int t = idxs[i]; idxs[i] = idxs[p]; idxs[p] = t;
        }
    }
}

// ---------------------------------------------------------------------------
// Gather top PRE_NMS candidates
// ---------------------------------------------------------------------------
__global__ void gather_cand(const float* __restrict__ keys, const int* __restrict__ idxs,
                            const float* __restrict__ boxes,
                            float* __restrict__ cand, float* __restrict__ vals)
{
    int i = blockIdx.x * blockDim.x + threadIdx.x;
    if (i >= PRE_NMS_) return;
    int s = idxs[i];
    cand[i * 4 + 0] = boxes[s * 4 + 0];
    cand[i * 4 + 1] = boxes[s * 4 + 1];
    cand[i * 4 + 2] = boxes[s * 4 + 2];
    cand[i * 4 + 3] = boxes[s * 4 + 3];
    vals[i] = keys[i];
}

// ---------------------------------------------------------------------------
// Greedy NMS, single workgroup, suppression flags in LDS
// ---------------------------------------------------------------------------
__device__ __forceinline__ float iou_one(float by1, float bx1, float by2, float bx2,
                                         const float* __restrict__ q)
{
    float ty = fmaxf(by1, q[0]);
    float tx = fmaxf(bx1, q[1]);
    float by = fminf(by2, q[2]);
    float bx = fminf(bx2, q[3]);
    float inter = fmaxf(by - ty, 0.0f) * fmaxf(bx - tx, 0.0f);
    float a1 = (by2 - by1) * (bx2 - bx1);
    float a2 = (q[2] - q[0]) * (q[3] - q[1]);
    return inter / (a1 + a2 - inter + 1e-9f);
}

__global__ void nms_kernel(const float* __restrict__ cand, const float* __restrict__ vals,
                           int* __restrict__ sel)
{
    __shared__ unsigned char sup[PRE_NMS_];
    __shared__ int cur_s;
    __shared__ int pick_s;
    __shared__ int alive_s;
    int tid = threadIdx.x;

    for (int i = tid; i < PRE_NMS_; i += blockDim.x)
        sup[i] = (vals[i] == -INFINITY) ? 1 : 0;
    if (tid == 0) cur_s = 0;
    __syncthreads();

    for (int k = 0; k < POST_NMS_; ++k) {
        if (tid == 0) {
            int c = cur_s;
            while (c < PRE_NMS_ && sup[c]) ++c;
            cur_s   = c;
            alive_s = (c < PRE_NMS_) ? 1 : 0;
            pick_s  = (c < PRE_NMS_) ? c : 0;   // argmax of all-false mask == 0
            sel[k]  = pick_s;
        }
        __syncthreads();
        if (alive_s) {
            int c = pick_s;
            float by1 = cand[c * 4 + 0], bx1 = cand[c * 4 + 1];
            float by2 = cand[c * 4 + 2], bx2 = cand[c * 4 + 3];
            for (int j = tid; j < PRE_NMS_; j += blockDim.x) {
                if (!sup[j]) {
                    float v = iou_one(by1, bx1, by2, bx2, cand + j * 4);
                    if (v > NMS_THRESH_) sup[j] = 1;
                }
            }
        }
        __syncthreads();
    }
}

// ---------------------------------------------------------------------------
// Gather selected rois -> workspace + output
// ---------------------------------------------------------------------------
__global__ void write_rois(const float* __restrict__ cand, const int* __restrict__ sel,
                           float* __restrict__ rois, float* __restrict__ out_rois)
{
    int k = blockIdx.x * blockDim.x + threadIdx.x;
    if (k >= POST_NMS_) return;
    int s = sel[k];
#pragma unroll
    for (int j = 0; j < 4; ++j) {
        float v = cand[s * 4 + j];
        rois[k * 4 + j] = v;
        out_rois[k * 4 + j] = v;
    }
}

// ---------------------------------------------------------------------------
// ROI adaptive 7x7 max-pool over [512, 37, 50] features
// ---------------------------------------------------------------------------
__global__ void roi_pool(const float* __restrict__ feat, const float* __restrict__ rois,
                         float* __restrict__ pooled)
{
    int roi = blockIdx.x;
    float r0 = rois[roi * 4 + 0], r1 = rois[roi * 4 + 1];
    float r2 = rois[roi * 4 + 2], r3 = rois[roi * 4 + 3];
    int ys = (int)floorf(r0 / 16.0f);
    int xs = (int)floorf(r1 / 16.0f);
    int ye = (int)floorf(r2 / 16.0f) + 1;
    int xe = (int)floorf(r3 / 16.0f) + 1;
    int Hr = max(min(ye, FH_) - ys, 1);
    int Wr = max(min(xe, FW_) - xs, 1);

    for (int t = threadIdx.x; t < 512 * 49; t += blockDim.x) {
        int ch = t / 49;
        int r  = t - ch * 49;
        int i  = r / 7;
        int j  = r - i * 7;
        int rs = ys + (i * Hr) / 7;
        int re = ys + ((i + 1) * Hr + 6) / 7;
        int cs = xs + (j * Wr) / 7;
        int ce = xs + ((j + 1) * Wr + 6) / 7;
        int rlo = max(rs, 0), rhi = min(re, FH_);
        int clo = max(cs, 0), chi = min(ce, FW_);
        float v = -1e30f;
        const float* fp = feat + (size_t)ch * (FH_ * FW_);
        for (int rr = rlo; rr < rhi; ++rr)
            for (int cc = clo; cc < chi; ++cc)
                v = fmaxf(v, fp[rr * FW_ + cc]);
        pooled[(size_t)roi * 25088 + ch * 49 + i * 7 + j] = v;
    }
}

// ---------------------------------------------------------------------------
// Host-side orchestration
// ---------------------------------------------------------------------------
extern "C" void kernel_launch(void* const* d_in, const int* in_sizes, int n_in,
                              void* d_out, int out_size, void* d_ws, size_t ws_size,
                              hipStream_t stream)
{
    (void)in_sizes; (void)n_in; (void)out_size; (void)ws_size;

    // ---- inputs ----
    const float* x = (const float*)d_in[0];
    const float* vw[13]; const float* vb[13];
    for (int i = 0; i < 13; ++i) { vw[i] = (const float*)d_in[1 + i]; vb[i] = (const float*)d_in[14 + i]; }
    const float* rpn_conv_w  = (const float*)d_in[27];
    const float* rpn_conv_b  = (const float*)d_in[28];
    const float* rpn_score_w = (const float*)d_in[29];
    const float* rpn_score_b = (const float*)d_in[30];
    const float* rpn_loc_w   = (const float*)d_in[31];
    const float* rpn_loc_b   = (const float*)d_in[32];
    const float* fc1_w = (const float*)d_in[33];
    const float* fc1_b = (const float*)d_in[34];
    const float* fc2_w = (const float*)d_in[35];
    const float* fc2_b = (const float*)d_in[36];
    const float* hl_w  = (const float*)d_in[37];
    const float* hl_b  = (const float*)d_in[38];
    const float* hs_w  = (const float*)d_in[39];
    const float* hs_b  = (const float*)d_in[40];

    // ---- workspace carve-out ----
    char*  base = (char*)d_ws;
    size_t off  = 0;
    auto carve = [&](size_t bytes) -> void* {
        void* p = base + off;
        off += (bytes + 255) & ~(size_t)255;
        return p;
    };
    float* bufA     = (float*)carve((size_t)64 * 600 * 800 * 4);
    float* bufB     = (float*)carve((size_t)64 * 600 * 800 * 4);
    float* rpnloc   = (float*)carve((size_t)36 * FH_ * FW_ * 4);
    float* rpnscore = (float*)carve((size_t)18 * FH_ * FW_ * 4);
    float* boxes    = (float*)carve((size_t)N_ANCH_TOT * 4 * 4);
    float* keys     = (float*)carve((size_t)SORT_N * 4);
    int*   idxs     = (int*)  carve((size_t)SORT_N * 4);
    float* cand     = (float*)carve((size_t)PRE_NMS_ * 4 * 4);
    float* vals     = (float*)carve((size_t)PRE_NMS_ * 4);
    int*   sel      = (int*)  carve((size_t)POST_NMS_ * 4);
    float* rois     = (float*)carve((size_t)POST_NMS_ * 4 * 4);
    float* pooled   = (float*)carve((size_t)POST_NMS_ * 25088 * 4);
    float* fc1o     = (float*)carve((size_t)POST_NMS_ * 4096 * 4);
    float* fc2o     = (float*)carve((size_t)POST_NMS_ * 4096 * 4);

    float* out      = (float*)d_out;
    float* out_locs   = out;                       // [300, 76]
    float* out_scores = out + 300 * 76;            // [300, 19]
    float* out_rois   = out + 300 * 76 + 300 * 19; // [300, 4]

    auto conv = [&](const float* in, const float* w, const float* b, float* o,
                    int Cin, int Cout, int H, int W, int ks, int relu) {
        int HW = H * W;
        dim3 grid((HW + 255) / 256, (Cout + 15) / 16);
        conv_wmma<<<grid, 128, 0, stream>>>(in, w, b, o, Cin, Cout, H, W, ks, relu);
    };
    auto pool = [&](const float* in, float* o, int C, int H, int W) {
        size_t total = (size_t)C * (H / 2) * (W / 2);
        maxpool2<<<(unsigned)((total + 255) / 256), 256, 0, stream>>>(in, o, C, H, W);
    };
    auto fc = [&](const float* A, const float* W, const float* b, float* o,
                  int M, int N, int K, int relu) {
        dim3 grid((N + 255) / 256, (M + 15) / 16);
        fc_wmma<<<grid, 128, 0, stream>>>(A, W, b, o, M, N, K, relu);
    };

    // ---- VGG16 backbone ----
    conv(x,    vw[0],  vb[0],  bufA, 3,   64,  600, 800, 3, 1);
    conv(bufA, vw[1],  vb[1],  bufB, 64,  64,  600, 800, 3, 1);
    pool(bufB, bufA, 64, 600, 800);
    conv(bufA, vw[2],  vb[2],  bufB, 64,  128, 300, 400, 3, 1);
    conv(bufB, vw[3],  vb[3],  bufA, 128, 128, 300, 400, 3, 1);
    pool(bufA, bufB, 128, 300, 400);
    conv(bufB, vw[4],  vb[4],  bufA, 128, 256, 150, 200, 3, 1);
    conv(bufA, vw[5],  vb[5],  bufB, 256, 256, 150, 200, 3, 1);
    conv(bufB, vw[6],  vb[6],  bufA, 256, 256, 150, 200, 3, 1);
    pool(bufA, bufB, 256, 150, 200);
    conv(bufB, vw[7],  vb[7],  bufA, 256, 512, 75, 100, 3, 1);
    conv(bufA, vw[8],  vb[8],  bufB, 512, 512, 75, 100, 3, 1);
    conv(bufB, vw[9],  vb[9],  bufA, 512, 512, 75, 100, 3, 1);
    pool(bufA, bufB, 512, 75, 100);
    conv(bufB, vw[10], vb[10], bufA, 512, 512, FH_, FW_, 3, 1);
    conv(bufA, vw[11], vb[11], bufB, 512, 512, FH_, FW_, 3, 1);
    conv(bufB, vw[12], vb[12], bufA, 512, 512, FH_, FW_, 3, 1);
    float* feat = bufA;  // [512, 37, 50]

    // ---- RPN ----
    conv(feat, rpn_conv_w,  rpn_conv_b,  bufB,     512, 512, FH_, FW_, 3, 1);
    conv(bufB, rpn_loc_w,   rpn_loc_b,   rpnloc,   512, 36,  FH_, FW_, 1, 0);
    conv(bufB, rpn_score_w, rpn_score_b, rpnscore, 512, 18,  FH_, FW_, 1, 0);

    // ---- proposals: decode -> sort -> NMS ----
    sort_init<<<SORT_N / 256, 256, 0, stream>>>(keys, idxs);
    proposal_prep<<<(N_ANCH_TOT + 255) / 256, 256, 0, stream>>>(rpnloc, rpnscore, boxes, keys);
    for (int k = 2; k <= SORT_N; k <<= 1)
        for (int j = k >> 1; j > 0; j >>= 1)
            bitonic_step<<<SORT_N / 256, 256, 0, stream>>>(keys, idxs, j, k);
    gather_cand<<<(PRE_NMS_ + 255) / 256, 256, 0, stream>>>(keys, idxs, boxes, cand, vals);
    nms_kernel<<<1, 256, 0, stream>>>(cand, vals, sel);
    write_rois<<<(POST_NMS_ + 255) / 256, 256, 0, stream>>>(cand, sel, rois, out_rois);

    // ---- ROI pooling + detection head ----
    roi_pool<<<POST_NMS_, 256, 0, stream>>>(feat, rois, pooled);
    fc(pooled, fc1_w, fc1_b, fc1o,       POST_NMS_, 4096, 25088, 1);
    fc(fc1o,   fc2_w, fc2_b, fc2o,       POST_NMS_, 4096, 4096,  1);
    fc(fc2o,   hl_w,  hl_b,  out_locs,   POST_NMS_, 76,   4096,  0);
    fc(fc2o,   hs_w,  hs_b,  out_scores, POST_NMS_, 19,   4096,  0);
}
